// BloomEmbed_10222022165187
// MI455X (gfx1250) — matrix operand order repaired and checked by reading, not compiled, not run on hardware
//
#include <hip/hip_runtime.h>
#include <stdint.h>

// BloomEmbed for gfx1250:
//   1) CSR-invert bloom_i (histogram + scan + bucket fill, ~400k atomics
//      instead of 25.7M), then agg rows built with plain loads/stores.
//   2) out = gelu(agg[tokens] @ w1 + b1) @ w2 + b2 via v_wmma_f32_16x16x32_bf16.
// GEMM1 computed transposed (h^T = w1^T * x^T): its C-layout output is
// bit-identical to GEMM2's A-fragment layout, so h never leaves VGPRs.
// GEMM1/GEMM2 interleaved per k-slice to overlap loads, WMMAs and GELU.

#define VOCAB  50257
#define EMBED  128
#define HIDDEN 512

typedef __attribute__((ext_vector_type(16))) __bf16 v16bf;
typedef __attribute__((ext_vector_type(8)))  float  v8f;

__device__ __forceinline__ unsigned short f2bf_bits(float x) {
    union { float f; uint32_t u; } c; c.f = x;
    uint32_t u = c.u;
    u += 0x7FFFu + ((u >> 16) & 1u);          // round-to-nearest-even
    return (unsigned short)(u >> 16);
}

// branchless tanh-GELU: x - x*rcp(exp(2u)+1), u = 0.79788456*(x+0.044715*x^3)
__device__ __forceinline__ float gelu_fast(float x) {
    float u  = 0.7978845608028654f * (x + 0.044715f * x * x * x);
    float e2 = __builtin_amdgcn_exp2f(u * 2.8853900817779268f);   // exp(2u)
    return x - x * __builtin_amdgcn_rcpf(e2 + 1.0f);
}

// ---------------------------------------------------------------------------
// CSR build of the bloom scatter
// ---------------------------------------------------------------------------
__global__ void hist_kernel(const int* __restrict__ bi, int* __restrict__ cnt, int n) {
    int e = blockIdx.x * blockDim.x + threadIdx.x;
    if (e < n) atomicAdd(&cnt[bi[e]], 1);
}

// single-block exclusive scan over cnt[VOCAB] -> offs[VOCAB+1]
__global__ void scan_kernel(const int* __restrict__ cnt, int* __restrict__ offs) {
    __shared__ int sh[1024];
    const int tid = threadIdx.x;
    int carry = 0;
    for (int base = 0; base < VOCAB; base += 1024) {
        int i = base + tid;
        int v = (i < VOCAB) ? cnt[i] : 0;
        sh[tid] = v;
        __syncthreads();
        for (int off = 1; off < 1024; off <<= 1) {
            int t = (tid >= off) ? sh[tid - off] : 0;
            __syncthreads();
            sh[tid] += t;
            __syncthreads();
        }
        if (i < VOCAB) offs[i] = carry + sh[tid] - v;   // exclusive
        carry += sh[1023];
        __syncthreads();
    }
    if (tid == 0) offs[VOCAB] = carry;
}

__global__ void fill_kernel(const int* __restrict__ bi, const int* __restrict__ bj,
                            const int* __restrict__ offs, int* __restrict__ fill,
                            int* __restrict__ slot, int n) {
    int e = blockIdx.x * blockDim.x + threadIdx.x;
    if (e >= n) return;
    int i = bi[e];
    int p = atomicAdd(&fill[i], 1);
    slot[offs[i] + p] = bj[e];
}

// agg[row] = 0.5 * sum(table[slot[s]]) over row's bucket; one wave per row
__global__ void build_agg(const float* __restrict__ table,
                          const int*   __restrict__ slot,
                          const int*   __restrict__ offs,
                          float*       __restrict__ agg) {
    int gid  = blockIdx.x * blockDim.x + threadIdx.x;
    int row  = gid >> 5;
    int lane = gid & 31;
    if (row >= VOCAB) return;
    int s0 = offs[row], s1 = offs[row + 1];
    float4 acc = {0.f, 0.f, 0.f, 0.f};
    for (int s = s0; s < s1; ++s) {
        int j = slot[s];
        const float4 v = *(const float4*)(table + (size_t)j * EMBED + lane * 4);
        acc.x += v.x; acc.y += v.y; acc.z += v.z; acc.w += v.w;
    }
    float4 r = {0.5f * acc.x, 0.5f * acc.y, 0.5f * acc.z, 0.5f * acc.w};
    *(float4*)(agg + (size_t)row * EMBED + lane * 4) = r;
}

// ---------------------------------------------------------------------------
// w1f: bf16 A-fragment order for w1^T tiles: [ntile(32)][ktile(4)][lane][e]
// A (16x32, MxK): M(row n) = lane&15, K = e + (lane>=16 ? 16 : 0)
// ---------------------------------------------------------------------------
__global__ void prep_w1(const float* __restrict__ w1, unsigned short* __restrict__ w1f) {
    int idx = blockIdx.x * blockDim.x + threadIdx.x;       // 32*4*32*16 = 65536
    if (idx >= 65536) return;
    int e    =  idx        & 15;
    int lane = (idx >> 4)  & 31;
    int kt   = (idx >> 9)  & 3;
    int nt   =  idx >> 11;
    int n = nt * 16 + (lane & 15);
    int K = kt * 32 + e + ((lane >= 16) ? 16 : 0);
    w1f[idx] = f2bf_bits(w1[K * HIDDEN + n]);
}

// w2f: bf16 B-fragment order: [ntile(8)][ktile(16)][lane][e]
// B (32x16, KxN): N = lane&15, K = (e<8 ? e : e+8) + (lane>=16 ? 8 : 0)
__global__ void prep_w2(const float* __restrict__ w2, unsigned short* __restrict__ w2f) {
    int idx = blockIdx.x * blockDim.x + threadIdx.x;       // 8*16*32*16 = 65536
    if (idx >= 65536) return;
    int e    =  idx        & 15;
    int lane = (idx >> 4)  & 31;
    int kt   = (idx >> 9)  & 15;
    int nt   =  idx >> 13;
    int N = nt * 16 + (lane & 15);
    int K = kt * 32 + ((e < 8) ? e : e + 8) + ((lane >= 16) ? 8 : 0);
    w2f[idx] = f2bf_bits(w2[K * EMBED + N]);
}

// ---------------------------------------------------------------------------
// Fused gather + MLP. 4 waves/block; each wave owns one 16-token group.
// ---------------------------------------------------------------------------
__launch_bounds__(128, 1)
__global__ void mlp_kernel(const int*            __restrict__ tokens,
                           const float*          __restrict__ agg,
                           const unsigned short* __restrict__ w1f,
                           const unsigned short* __restrict__ w2f,
                           const float*          __restrict__ b1,
                           const float*          __restrict__ b2,
                           float*                __restrict__ out) {
    const int lane  = threadIdx.x & 31;
    const int wave  = threadIdx.x >> 5;
    const int group = blockIdx.x * 4 + wave;               // 16-token group id
    const int m     = lane & 15;
    const int hi    = lane >> 4;                           // 0 or 1

    // --- x^T B-fragments: lane = token column m, K = kt*32 + hi*16 + e (contiguous)
    const int tok = tokens[group * 16 + m];
    const float* xrow = agg + (size_t)tok * EMBED;
    v16bf xfrag[4];
#pragma unroll
    for (int kt = 0; kt < 4; ++kt) {
        const float* p = xrow + kt * 32 + hi * 16;
        float4 p0 = *(const float4*)(p + 0);
        float4 p1 = *(const float4*)(p + 4);
        float4 p2 = *(const float4*)(p + 8);
        float4 p3 = *(const float4*)(p + 12);
        union { v16bf v; __bf16 s[16]; } u;
        u.s[0]  = (__bf16)p0.x; u.s[1]  = (__bf16)p0.y; u.s[2]  = (__bf16)p0.z; u.s[3]  = (__bf16)p0.w;
        u.s[4]  = (__bf16)p1.x; u.s[5]  = (__bf16)p1.y; u.s[6]  = (__bf16)p1.z; u.s[7]  = (__bf16)p1.w;
        u.s[8]  = (__bf16)p2.x; u.s[9]  = (__bf16)p2.y; u.s[10] = (__bf16)p2.z; u.s[11] = (__bf16)p2.w;
        u.s[12] = (__bf16)p3.x; u.s[13] = (__bf16)p3.y; u.s[14] = (__bf16)p3.z; u.s[15] = (__bf16)p3.w;
        xfrag[kt] = u.v;
    }

    // --- GEMM2 accumulators (bias pre-loaded); filled as GEMM1 k-slices arrive
    v8f acc2[8];
#pragma unroll
    for (int nt2 = 0; nt2 < 8; ++nt2) {
        const float bias = b2[nt2 * 16 + m];
#pragma unroll
        for (int r = 0; r < 8; ++r) acc2[nt2][r] = bias;
    }

    // --- interleaved GEMM1 (transposed, +GELU) and GEMM2 per 32-wide k-slice
#pragma unroll
    for (int kt2 = 0; kt2 < HIDDEN / 32; ++kt2) {
        union { v16bf v; __bf16 s[16]; } hfr;              // one GEMM2 A fragment
#pragma unroll
        for (int half = 0; half < 2; ++half) {
            const int nt = kt2 * 2 + half;
            const float4 bb0 = *(const float4*)(b1 + nt * 16 + hi * 8);
            const float4 bb1 = *(const float4*)(b1 + nt * 16 + hi * 8 + 4);
            v8f acc;
            acc[0] = bb0.x; acc[1] = bb0.y; acc[2] = bb0.z; acc[3] = bb0.w;
            acc[4] = bb1.x; acc[5] = bb1.y; acc[6] = bb1.z; acc[7] = bb1.w;
#pragma unroll
            for (int kt = 0; kt < 4; ++kt) {
                const v16bf afr = *(const v16bf*)(w1f + (size_t)((nt * 4 + kt) * 32 + lane) * 16);
                acc = __builtin_amdgcn_wmma_f32_16x16x32_bf16(
                    false, afr, false, xfrag[kt], (short)0, acc, false, false);
            }
#pragma unroll
            for (int r = 0; r < 8; ++r)
                hfr.s[half * 8 + r] = (__bf16)gelu_fast(acc[r]);
        }
#pragma unroll
        for (int nt2 = 0; nt2 < 8; ++nt2) {
            const v16bf bfr = *(const v16bf*)(w2f + (size_t)((nt2 * 16 + kt2) * 32 + lane) * 16);
            acc2[nt2] = __builtin_amdgcn_wmma_f32_16x16x32_bf16(
                false, hfr.v, false, bfr, (short)0, acc2[nt2], false, false);
        }
    }

    // --- store out[16,128]
#pragma unroll
    for (int nt2 = 0; nt2 < 8; ++nt2) {
        float* orow = out + ((size_t)group * 16 + hi * 8) * EMBED + nt2 * 16 + m;
#pragma unroll
        for (int r = 0; r < 8; ++r)
            orow[(size_t)r * EMBED] = acc2[nt2][r];
    }
}

// ---------------------------------------------------------------------------
extern "C" void kernel_launch(void* const* d_in, const int* in_sizes, int n_in,
                              void* d_out, int out_size, void* d_ws, size_t ws_size,
                              hipStream_t stream) {
    const int*   tokens = (const int*)  d_in[0];
    const float* table  = (const float*)d_in[1];
    const int*   bi     = (const int*)  d_in[2];
    const int*   bj     = (const int*)  d_in[3];
    const float* w1     = (const float*)d_in[4];
    const float* b1     = (const float*)d_in[5];
    const float* w2     = (const float*)d_in[6];
    const float* b2     = (const float*)d_in[7];
    float* out = (float*)d_out;

    const int nEntries = in_sizes[2];   // NUM_DIGEST * VOCAB = 201028
    const int nTok     = in_sizes[0];   // 8 * 4096 = 32768

    // workspace layout (256B aligned chunks)
    auto align256 = [](size_t x) { return (x + 255) & ~(size_t)255; };
    char* ws = (char*)d_ws;
    size_t o = 0;
    float* agg = (float*)(ws + o);          o += align256((size_t)VOCAB * EMBED * sizeof(float));
    unsigned short* w1f = (unsigned short*)(ws + o); o += align256(65536 * sizeof(unsigned short));
    unsigned short* w2f = (unsigned short*)(ws + o); o += align256(65536 * sizeof(unsigned short));
    int* cnt  = (int*)(ws + o);             o += align256((size_t)VOCAB * sizeof(int));
    int* fill = (int*)(ws + o);             o += align256((size_t)VOCAB * sizeof(int));
    int* offs = (int*)(ws + o);             o += align256((size_t)(VOCAB + 1) * sizeof(int));
    int* slot = (int*)(ws + o);             o += align256((size_t)nEntries * sizeof(int));

    // zero the two counter arrays (cnt, fill are adjacent-ish; two memsets)
    hipMemsetAsync(cnt,  0, (size_t)VOCAB * sizeof(int), stream);
    hipMemsetAsync(fill, 0, (size_t)VOCAB * sizeof(int), stream);

    const int eb = (nEntries + 255) / 256;
    hist_kernel<<<eb, 256, 0, stream>>>(bi, cnt, nEntries);
    scan_kernel<<<1, 1024, 0, stream>>>(cnt, offs);
    fill_kernel<<<eb, 256, 0, stream>>>(bi, bj, offs, fill, slot, nEntries);
    build_agg<<<((size_t)VOCAB * 32 + 255) / 256, 256, 0, stream>>>(table, slot, offs, agg);

    prep_w1<<<65536 / 256, 256, 0, stream>>>(w1, w1f);
    prep_w2<<<65536 / 256, 256, 0, stream>>>(w2, w2f);

    mlp_kernel<<<nTok / 64, 128, 0, stream>>>(tokens, agg, w1f, w2f, b1, b2, out);
}
